// TrmBlock_49168785604833
// MI455X (gfx1250) — compile-verified, matrix operand order checked by reading, not run on hardware
//
#include <hip/hip_runtime.h>
#include <hip/hip_bf16.h>

// ---------------------------------------------------------------------------
// Transformer encoder block on gfx1250 (MI455X), bf16 WMMA everywhere.
//   x:[4,2048,512] f32 -> out f32.  All matmuls: v_wmma_f32_16x16x32_bf16.
//   - GEMMs: weight tiles staged block-wide through LDS with async-to-LDS
//     copies (ASYNCcnt), double-buffered so WMMAs overlap the next fetch.
//   - Attention is flash-style (online softmax): the 512MB score tensor is
//     never materialized; tiles live in registers + 1KB LDS per wave, with
//     global_prefetch warming the next K/V tiles.
// ---------------------------------------------------------------------------

typedef __attribute__((ext_vector_type(16))) __bf16 v16bf;
typedef __attribute__((ext_vector_type(8)))  float  v8f;

#define DMODEL 512
#define LSEQ   2048
#define NBATCH 4
#define NHEAD  8
#define EHEAD  64
#define NTOK   (NBATCH * LSEQ)   // 8192 token rows

static __device__ __forceinline__ v8f wmma_bf16(v16bf a, v16bf b, v8f c) {
  // D(f32 16x16) = A(bf16 16x32) * B(bf16 32x16) + C
  return __builtin_amdgcn_wmma_f32_16x16x32_bf16(false, a, false, b,
                                                 (short)0, c, false, false);
}

// Async DMA of 16 bytes global -> LDS (tracked by ASYNCcnt, no VGPR bounce).
static __device__ __forceinline__ void async_b128_to_lds(unsigned int lds_off,
                                                         const __bf16* gaddr) {
  asm volatile("global_load_async_to_lds_b128 %0, %1, off"
               :: "v"(lds_off), "v"(gaddr)
               : "memory");
}

// --------------------------- fp32 -> bf16 convert ---------------------------
__global__ __launch_bounds__(256)
void k_f32_to_bf16(const float* __restrict__ in, __bf16* __restrict__ out, int n) {
  int i = blockIdx.x * 256 + threadIdx.x;
  const int stride = gridDim.x * 256;
  for (; i < n; i += stride) out[i] = (__bf16)in[i];
}

// --------------------------- generic WMMA GEMM ------------------------------
// Y[M,N] = A[M,K] @ W[N,K]^T + bias ;  M = NTOK, N = gridDim.x*64.
// block = (32,4): 4 waves, each wave owns a 16(M) x 64(N) tile. The 64x32
// weight k-tile (4KB) is shared by all 4 waves: staged once per block via
// async-to-LDS, double-buffered (issue next tile, s_wait_asynccnt 0x2, use
// current). A-operands are register double-buffered the same way.
// mode: 0 = store bf16, 1 = store bf16 with ReLU, 2 = store f32.
__global__ __launch_bounds__(128)
void k_gemm_wmma(const __bf16* __restrict__ A, const __bf16* __restrict__ W,
                 const float* __restrict__ bias,
                 __bf16* __restrict__ outb, float* __restrict__ outf,
                 int K, int mode) {
  const int tid    = threadIdx.y * 32 + threadIdx.x;
  const int lane16 = threadIdx.x & 15;
  const int hl     = threadIdx.x >> 4;
  const int m0 = (blockIdx.y * 4 + threadIdx.y) * 16;
  const int n0 = blockIdx.x * 64;
  const int N  = gridDim.x * 64;

  __shared__ __attribute__((aligned(128))) __bf16 lds_w[2][64 * 32];

  // Cooperative staging: 128 threads x 32B = 4KB tile. Thread covers
  // W row n = tid>>1, k-halfwave (tid&1)*16 (two b128 async copies).
  const int wn = tid >> 1;
  const int wk = (tid & 1) * 16;
  const __bf16* wsrc = W + (size_t)(n0 + wn) * K + wk;
  const unsigned int ldst0 =
      (unsigned int)(uintptr_t)(&lds_w[0][wn * 32 + wk]);
  const unsigned int ldst1 =
      (unsigned int)(uintptr_t)(&lds_w[1][wn * 32 + wk]);

  // prologue: stage k-tile 0 into buffer 0
  async_b128_to_lds(ldst0, wsrc);
  async_b128_to_lds(ldst0 + 16, wsrc + 8);

  v8f c[4] = {v8f{}, v8f{}, v8f{}, v8f{}};
  const __bf16* arow = A + (size_t)(m0 + lane16) * K + hl * 16;
  v16bf a_cur = *(const v16bf*)(arow);
  v16bf a_nxt;

  for (int k0 = 0; k0 < K; k0 += 32) {
    const int cur = (k0 >> 5) & 1;
    const bool has_next = (k0 + 32) < K;
    if (has_next) {
      // issue next weight tile into the other buffer, keep it in flight
      const unsigned int nd = cur ? ldst0 : ldst1;
      async_b128_to_lds(nd, wsrc + k0 + 32);
      async_b128_to_lds(nd + 16, wsrc + k0 + 40);
      a_nxt = *(const v16bf*)(arow + k0 + 32);         // overlap A fetch
      asm volatile("s_wait_asynccnt 0x2" ::: "memory"); // cur done, next inflight
    } else {
      asm volatile("s_wait_asynccnt 0x0" ::: "memory");
    }
    __syncthreads();   // all waves' async stores to lds_w[cur] visible

    const __bf16* lw = &lds_w[cur][0];
    v16bf b0 = *(const v16bf*)(lw + (0 * 16 + lane16) * 32 + hl * 16);
    v16bf b1 = *(const v16bf*)(lw + (1 * 16 + lane16) * 32 + hl * 16);
    v16bf b2 = *(const v16bf*)(lw + (2 * 16 + lane16) * 32 + hl * 16);
    v16bf b3 = *(const v16bf*)(lw + (3 * 16 + lane16) * 32 + hl * 16);
    c[0] = wmma_bf16(a_cur, b0, c[0]);
    c[1] = wmma_bf16(a_cur, b1, c[1]);
    c[2] = wmma_bf16(a_cur, b2, c[2]);
    c[3] = wmma_bf16(a_cur, b3, c[3]);

    __syncthreads();   // done reading lds_w[cur]; next iter may overwrite it
    a_cur = a_nxt;
  }

#pragma unroll
  for (int j = 0; j < 4; ++j) {
    const int col = n0 + j * 16 + lane16;
    const float bv = bias[col];
#pragma unroll
    for (int i = 0; i < 8; ++i) {
      const int row = m0 + hl * 8 + i;        // C layout: lanes>=16 hold M=8+i
      float v = c[j][i] + bv;
      if (mode == 1) v = fmaxf(v, 0.0f);
      if (mode == 2) outf[(size_t)row * N + col] = v;
      else           outb[(size_t)row * N + col] = (__bf16)v;
    }
  }
}

// ------------------- V head-transpose: [B*L,512] -> [B*H][E][L] -------------
__global__ __launch_bounds__(256)
void k_transpose_v(const __bf16* __restrict__ V, __bf16* __restrict__ Vt) {
  const int i = blockIdx.x * 256 + threadIdx.x;   // over 4*8*64*2048 = 2^22
  const int l  = i & (LSEQ - 1);
  const int e  = (i >> 11) & (EHEAD - 1);
  const int bh = i >> 17;
  const int b = bh >> 3, h = bh & 7;
  Vt[i] = V[(size_t)(b * LSEQ + l) * DMODEL + h * EHEAD + e];
}

// --------------------------- flash attention --------------------------------
// grid = (L/16, B*H), block = 32 (one wave). Wave owns a 16-query tile:
//   per 32-key step: S(16x32) = Q(16x64) Kt -> 4 WMMAs, online softmax via
//   shfl_xor reductions inside each 16-lane half, P re-staged through LDS into
//   A-operand layout, O += P(16x32) V(32x64) -> 4 WMMAs. Next K/V tiles are
//   prefetched so the softmax VALU work overlaps the fetch.
__global__ __launch_bounds__(32)
void k_attn_wmma(const __bf16* __restrict__ Q, const __bf16* __restrict__ Kb,
                 const __bf16* __restrict__ Vt, __bf16* __restrict__ Ctx) {
  const int lane16 = threadIdx.x & 15;
  const int hl     = threadIdx.x >> 4;
  const int q0 = blockIdx.x * 16;
  const int bh = blockIdx.y;
  const int b = bh >> 3, h = bh & 7;
  const size_t tok0 = (size_t)b * LSEQ;
  const float kScale = 0.125f;                 // 1/sqrt(E), E = 64

  __shared__ __attribute__((aligned(64))) __bf16 lds_p[16 * 32];

  // Q tile as two 16x32 A-operands (E chunks 0..31, 32..63)
  const __bf16* qrow = Q + (tok0 + q0 + lane16) * DMODEL + h * EHEAD + hl * 16;
  v16bf qa0 = *(const v16bf*)(qrow);
  v16bf qa1 = *(const v16bf*)(qrow + 32);

  v8f o[4] = {v8f{}, v8f{}, v8f{}, v8f{}};     // O tile 16 x 64
  float mrow[8], lrow[8];
#pragma unroll
  for (int i = 0; i < 8; ++i) { mrow[i] = -1e30f; lrow[i] = 0.0f; }

  const __bf16* Vth = Vt + (size_t)bh * EHEAD * LSEQ;

  for (int kt = 0; kt < LSEQ; kt += 32) {
    if (kt + 32 < LSEQ) {   // warm next K rows and V columns (global_prefetch)
      __builtin_prefetch(Kb + (tok0 + kt + 32 + threadIdx.x) * DMODEL + h * EHEAD,
                         0, 3);
      __builtin_prefetch(Vth + (size_t)(threadIdx.x << 1) * LSEQ + kt + 32, 0, 3);
    }

    // ---- S = scale * Q K^T, two 16x16 tiles (keys kt..+15, kt+16..+31) ----
    v8f s[2] = {v8f{}, v8f{}};
#pragma unroll
    for (int t = 0; t < 2; ++t) {
      const __bf16* krow =
          Kb + (tok0 + kt + t * 16 + lane16) * DMODEL + h * EHEAD + hl * 16;
      v16bf kb0 = *(const v16bf*)(krow);
      v16bf kb1 = *(const v16bf*)(krow + 32);
      s[t] = wmma_bf16(qa0, kb0, s[t]);
      s[t] = wmma_bf16(qa1, kb1, s[t]);
    }

    // ---- online softmax update (row stats per C-component) ----
#pragma unroll
    for (int i = 0; i < 8; ++i) {
      float x0 = s[0][i] * kScale;
      float x1 = s[1][i] * kScale;
      float r = fmaxf(x0, x1);                 // reduce over N within half
      r = fmaxf(r, __shfl_xor(r, 1, 32));
      r = fmaxf(r, __shfl_xor(r, 2, 32));
      r = fmaxf(r, __shfl_xor(r, 4, 32));
      r = fmaxf(r, __shfl_xor(r, 8, 32));
      const float mn   = fmaxf(mrow[i], r);
      const float corr = __expf(mrow[i] - mn);
      const float p0 = __expf(x0 - mn);
      const float p1 = __expf(x1 - mn);
      float rs = p0 + p1;
      rs += __shfl_xor(rs, 1, 32);
      rs += __shfl_xor(rs, 2, 32);
      rs += __shfl_xor(rs, 4, 32);
      rs += __shfl_xor(rs, 8, 32);
      lrow[i] = lrow[i] * corr + rs;
      mrow[i] = mn;
      o[0][i] = o[0][i] * corr;
      o[1][i] = o[1][i] * corr;
      o[2][i] = o[2][i] * corr;
      o[3][i] = o[3][i] * corr;
      lds_p[(hl * 8 + i) * 32 + lane16]      = (__bf16)p0;
      lds_p[(hl * 8 + i) * 32 + 16 + lane16] = (__bf16)p1;
    }
    __syncthreads();                            // single-wave WG: S_NOP
    v16bf pa = *(const v16bf*)(&lds_p[lane16 * 32 + hl * 16]);
    __syncthreads();

    // ---- O += P(16x32) @ V(32x64): B-tile rows contiguous in Vt ----
#pragma unroll
    for (int j = 0; j < 4; ++j) {
      const __bf16* vrow = Vth + (size_t)(j * 16 + lane16) * LSEQ + kt + hl * 16;
      v16bf vb = *(const v16bf*)(vrow);
      o[j] = wmma_bf16(pa, vb, o[j]);
    }
  }

  // ---- normalize and write context (bf16) ----
#pragma unroll
  for (int i = 0; i < 8; ++i) {
    const float inv = 1.0f / lrow[i];
    __bf16* crow = Ctx + (tok0 + q0 + hl * 8 + i) * DMODEL + h * EHEAD;
#pragma unroll
    for (int j = 0; j < 4; ++j)
      crow[j * 16 + lane16] = (__bf16)(o[j][i] * inv);
  }
}

// ------------------- residual + LayerNorm (row of 512 per block) ------------
__global__ __launch_bounds__(256)
void k_residual_ln(const float* __restrict__ A, const float* __restrict__ B,
                   const float* __restrict__ gamma, const float* __restrict__ beta,
                   float* __restrict__ outf, __bf16* __restrict__ outb) {
  const int row = blockIdx.x;
  const int tid = threadIdx.x;
  const size_t base = (size_t)row * DMODEL;
  const float v0 = A[base + tid]       + B[base + tid];
  const float v1 = A[base + tid + 256] + B[base + tid + 256];
  float s  = v0 + v1;
  float sq = v0 * v0 + v1 * v1;
#pragma unroll
  for (int m = 1; m < 32; m <<= 1) {
    s  += __shfl_xor(s,  m, 32);
    sq += __shfl_xor(sq, m, 32);
  }
  __shared__ float red_s[8], red_q[8];
  const int wid = tid >> 5;
  if ((tid & 31) == 0) { red_s[wid] = s; red_q[wid] = sq; }
  __syncthreads();
  if (tid < 32) {
    float ts = (tid < 8) ? red_s[tid] : 0.0f;
    float tq = (tid < 8) ? red_q[tid] : 0.0f;
#pragma unroll
    for (int m = 1; m < 8; m <<= 1) {
      ts += __shfl_xor(ts, m, 32);
      tq += __shfl_xor(tq, m, 32);
    }
    if (tid == 0) { red_s[0] = ts; red_q[0] = tq; }
  }
  __syncthreads();
  const float mean = red_s[0] * (1.0f / DMODEL);
  const float var  = red_q[0] * (1.0f / DMODEL) - mean * mean;
  const float rstd = rsqrtf(var + 1e-5f);
  const float r0 = (v0 - mean) * rstd * gamma[tid]       + beta[tid];
  const float r1 = (v1 - mean) * rstd * gamma[tid + 256] + beta[tid + 256];
  outf[base + tid]       = r0;
  outf[base + tid + 256] = r1;
  if (outb) {
    outb[base + tid]       = (__bf16)r0;
    outb[base + tid + 256] = (__bf16)r1;
  }
}

// ------------------------------- launcher -----------------------------------
extern "C" void kernel_launch(void* const* d_in, const int* in_sizes, int n_in,
                              void* d_out, int out_size, void* d_ws, size_t ws_size,
                              hipStream_t stream) {
  (void)in_sizes; (void)n_in; (void)out_size; (void)ws_size;
  const float* x  = (const float*)d_in[0];
  const float* Wq = (const float*)d_in[1];
  const float* bq = (const float*)d_in[2];
  const float* Wk = (const float*)d_in[3];
  const float* bk = (const float*)d_in[4];
  const float* Wv = (const float*)d_in[5];
  const float* bv = (const float*)d_in[6];
  const float* Wo = (const float*)d_in[7];
  const float* bo = (const float*)d_in[8];
  const float* g1 = (const float*)d_in[9];
  const float* be1= (const float*)d_in[10];
  const float* g2 = (const float*)d_in[11];
  const float* be2= (const float*)d_in[12];
  const float* W1 = (const float*)d_in[13];
  const float* b1 = (const float*)d_in[14];
  const float* W2 = (const float*)d_in[15];
  const float* b2 = (const float*)d_in[16];

  // -------- workspace carve-out (~83 MB, 256B aligned, with reuse) --------
  const size_t ACT_BF = (size_t)NTOK * DMODEL * sizeof(__bf16);   // 8 MB
  const size_t WT_BF  = (size_t)DMODEL * DMODEL * sizeof(__bf16); // 512 KB
  const size_t ACT_F  = (size_t)NTOK * DMODEL * sizeof(float);    // 16 MB
  char* ws = (char*)d_ws;
  size_t off = 0;
  auto take = [&](size_t bytes) {
    char* p = ws + off;
    off += (bytes + 255) & ~(size_t)255;
    return p;
  };
  __bf16* Xbf  = (__bf16*)take(ACT_BF);   // later reused as ff1 activations
  __bf16* Wqb  = (__bf16*)take(WT_BF);
  __bf16* Wkb  = (__bf16*)take(WT_BF);
  __bf16* Wvb  = (__bf16*)take(WT_BF);
  __bf16* Wob  = (__bf16*)take(WT_BF);
  __bf16* W1b  = (__bf16*)take(WT_BF);
  __bf16* W2b  = (__bf16*)take(WT_BF);
  __bf16* Qbf  = (__bf16*)take(ACT_BF);
  __bf16* Kbf  = (__bf16*)take(ACT_BF);
  __bf16* Vbf  = (__bf16*)take(ACT_BF);   // reused as attention context
  __bf16* Vt   = (__bf16*)take(ACT_BF);
  float*  attf = (float*)take(ACT_F);     // reused as ff2 output
  float*  hbuf = (float*)take(ACT_F);
  __bf16* hbf  = (__bf16*)take(ACT_BF);
  __bf16* Ctx  = Vbf;
  __bf16* ff1  = Xbf;
  float*  ff2f = attf;

  // -------- 1) bf16 conversions --------
  k_f32_to_bf16<<<4096, 256, 0, stream>>>(x,  Xbf, NTOK * DMODEL);
  k_f32_to_bf16<<<1024, 256, 0, stream>>>(Wq, Wqb, DMODEL * DMODEL);
  k_f32_to_bf16<<<1024, 256, 0, stream>>>(Wk, Wkb, DMODEL * DMODEL);
  k_f32_to_bf16<<<1024, 256, 0, stream>>>(Wv, Wvb, DMODEL * DMODEL);
  k_f32_to_bf16<<<1024, 256, 0, stream>>>(Wo, Wob, DMODEL * DMODEL);
  k_f32_to_bf16<<<1024, 256, 0, stream>>>(W1, W1b, DMODEL * DMODEL);
  k_f32_to_bf16<<<1024, 256, 0, stream>>>(W2, W2b, DMODEL * DMODEL);

  // -------- 2) Q/K/V projections --------
  const dim3 gblk(32, 4);
  const dim3 ggrd(DMODEL / 64, NTOK / 64);
  k_gemm_wmma<<<ggrd, gblk, 0, stream>>>(Xbf, Wqb, bq, Qbf, nullptr, DMODEL, 0);
  k_gemm_wmma<<<ggrd, gblk, 0, stream>>>(Xbf, Wkb, bk, Kbf, nullptr, DMODEL, 0);
  k_gemm_wmma<<<ggrd, gblk, 0, stream>>>(Xbf, Wvb, bv, Vbf, nullptr, DMODEL, 0);

  // -------- 3) per-head V transpose, 4) flash attention --------
  k_transpose_v<<<(NBATCH * NHEAD * EHEAD * LSEQ) / 256, 256, 0, stream>>>(Vbf, Vt);
  k_attn_wmma<<<dim3(LSEQ / 16, NBATCH * NHEAD), 32, 0, stream>>>(Qbf, Kbf, Vt, Ctx);

  // -------- 5) output projection + LN1 --------
  k_gemm_wmma<<<ggrd, gblk, 0, stream>>>(Ctx, Wob, bo, nullptr, attf, DMODEL, 2);
  k_residual_ln<<<NTOK, 256, 0, stream>>>(x, attf, g1, be1, hbuf, hbf);

  // -------- 6) FFN + LN2 --------
  k_gemm_wmma<<<ggrd, gblk, 0, stream>>>(hbf, W1b, b1, ff1, nullptr, DMODEL, 1);
  k_gemm_wmma<<<ggrd, gblk, 0, stream>>>(ff1, W2b, b2, nullptr, ff2f, DMODEL, 2);
  k_residual_ln<<<NTOK, 256, 0, stream>>>(hbuf, ff2f, g2, be2, (float*)d_out, nullptr);
}
